// HomoConv_19490561589642
// MI455X (gfx1250) — matrix operand order repaired and checked by reading, not compiled, not run on hardware
//
#include <hip/hip_runtime.h>

#define HDIM 512
#define LROW 72   // padded LDS row stride (elements) for a 64-wide K tile

typedef __attribute__((ext_vector_type(16))) __bf16        v16bf;
typedef __attribute__((ext_vector_type(8)))  float         v8f;
typedef __attribute__((ext_vector_type(4)))  unsigned int  v4u;

__device__ __forceinline__ unsigned short f2bf(float f) {
    unsigned int u = __float_as_uint(f);
    u += 0x7FFFu + ((u >> 16) & 1u);   // round-to-nearest-even
    return (unsigned short)(u >> 16);
}

// ---------------- scatter-add of edge features onto destination nodes --------
__global__ __launch_bounds__(256) void scatter_add_kernel(
        const float* __restrict__ ef, const int* __restrict__ dstIdx,
        float* __restrict__ msg, int E) {
    int idx = blockIdx.x * 256 + threadIdx.x;   // one thread = 4 columns
    int e = idx >> 7;
    if (e >= E) return;
    int c = (idx & 127) << 2;
    const float4 v = *(const float4*)(ef + (size_t)e * HDIM + c);
    float* m = msg + (size_t)dstIdx[e] * HDIM + c;
    atomicAdd(m + 0, v.x); atomicAdd(m + 1, v.y);
    atomicAdd(m + 2, v.z); atomicAdd(m + 3, v.w);
}

// ------- f32 [K,512] -> bf16 transposed [512,K] (LDS-tiled, coalesced) -------
__global__ __launch_bounds__(256) void cvt_transpose_kernel(
        const float* __restrict__ src, unsigned short* __restrict__ dst, int K) {
    __shared__ float t[32][33];
    const int k0 = blockIdx.x * 32, n0 = blockIdx.y * 32;
    const int tx = threadIdx.x & 31, ty = threadIdx.x >> 5;   // 32 x 8
    #pragma unroll
    for (int i = 0; i < 32; i += 8)
        t[ty + i][tx] = src[(size_t)(k0 + ty + i) * HDIM + n0 + tx];
    __syncthreads();
    #pragma unroll
    for (int i = 0; i < 32; i += 8)
        dst[(size_t)(n0 + ty + i) * K + k0 + tx] = f2bf(t[tx][ty + i]);
}

// ---------------- row LayerNorm + ReLU, emit bf16 ----------------------------
__global__ __launch_bounds__(256) void ln_relu_kernel(
        const float* __restrict__ in, const float* __restrict__ g,
        const float* __restrict__ be, unsigned short* __restrict__ out) {
    int row = blockIdx.x;
    int tid = threadIdx.x;                       // 256 threads, 2 cols each
    const float2 v = *(const float2*)(in + (size_t)row * HDIM + tid * 2);
    float s = v.x + v.y;
    float q = v.x * v.x + v.y * v.y;
    #pragma unroll
    for (int o = 16; o > 0; o >>= 1) {           // wave32 reduction
        s += __shfl_xor(s, o, 32);
        q += __shfl_xor(q, o, 32);
    }
    __shared__ float red[16];
    int wave = tid >> 5, lane = tid & 31;
    if (lane == 0) { red[wave] = s; red[wave + 8] = q; }
    __syncthreads();
    float ts = 0.f, tq = 0.f;
    #pragma unroll
    for (int i = 0; i < 8; i++) { ts += red[i]; tq += red[i + 8]; }
    float mu  = ts * (1.f / HDIM);
    float var = tq * (1.f / HDIM) - mu * mu;
    float rs  = rsqrtf(var + 1e-5f);
    int c = tid * 2;
    float o0 = fmaxf((v.x - mu) * rs * g[c]     + be[c],     0.f);
    float o1 = fmaxf((v.y - mu) * rs * g[c + 1] + be[c + 1], 0.f);
    unsigned short* op = out + (size_t)row * HDIM + c;
    op[0] = f2bf(o0); op[1] = f2bf(o1);
}

// ---------------- WMMA GEMM: C[M,512] = A[M,K] @ W[K,512] (+bias)(+resid) ----
// W supplied pre-transposed bf16: Wt[512][K].
// Block tile: 128(M) x 128(N), K-step 64, double-buffered LDS (padded rows).
// Grid: x = n-tile (4), y = m-tile  -> n-tiles of one row-block run adjacently
// so A rows + weights stay L2-resident.
// MODE 0: A is preconverted bf16 [M,K]
// MODE 1: A row r = concat(S0[idx0[r]], S0[idx1[r]], S1[r])  (K = 1536, f32 src)
// MODE 2: A row r = concat(S0[r], S1[r])                     (K = 1024, f32 src)
template <int MODE>
__global__ __launch_bounds__(256) void gemm512_kernel(
        const unsigned short* __restrict__ Abf,
        const float* __restrict__ S0, const float* __restrict__ S1,
        const int* __restrict__ idx0, const int* __restrict__ idx1,
        const unsigned short* __restrict__ Wtbf,
        const float* __restrict__ bias,
        const float* __restrict__ resid,
        float* __restrict__ Cout,
        int M, int K) {
    __shared__ unsigned short ldsA[2][128 * LROW];   // A tile [m=128][k=64] padded
    __shared__ unsigned short ldsB[2][128 * LROW];   // B tile transposed [n=128][k=64]

    const int tid  = threadIdx.x;
    const int wave = tid >> 5;
    const int lane = tid & 31;
    const int n0   = blockIdx.x * 128;
    const int m0   = blockIdx.y * 128;

    v8f acc[8];
    const v8f zero8 = {0.f, 0.f, 0.f, 0.f, 0.f, 0.f, 0.f, 0.f};
    #pragma unroll
    for (int i = 0; i < 8; i++) acc[i] = zero8;

    const int  arow  = tid >> 1;        // 0..127
    const int  acseg = (tid & 1) * 32;  // 0 or 32
    const int  grow  = m0 + arow;
    const bool rowOK = grow < M;
    int g0 = 0, g1 = 0;
    if (MODE == 1 && rowOK) { g0 = idx0[grow]; g1 = idx1[grow]; }

    const int bn    = tid >> 1;         // 0..127 (n within tile)
    const int bkseg = (tid & 1) * 32;   // 0 or 32 (k segment)

    union AU { v4u u[4]; unsigned short s[32]; };

    auto gather16 = [&](int c, unsigned short* s) {   // MODE 1/2 only
        const int off = c & (HDIM - 1);
        const float* src;
        if (MODE == 1) {
            const int qq = c >> 9;
            src = (qq == 0) ? (S0 + (size_t)g0 * HDIM + off)
                : (qq == 1) ? (S0 + (size_t)g1 * HDIM + off)
                            : (S1 + (size_t)grow * HDIM + off);
        } else {
            src = (c < HDIM) ? (S0 + (size_t)grow * HDIM + off)
                             : (S1 + (size_t)grow * HDIM + off);
        }
        #pragma unroll
        for (int i = 0; i < 4; i++) {
            const float4 f = *(const float4*)(src + i * 4);
            s[i * 4 + 0] = f2bf(f.x); s[i * 4 + 1] = f2bf(f.y);
            s[i * 4 + 2] = f2bf(f.z); s[i * 4 + 3] = f2bf(f.w);
        }
    };

    auto loadAB = [&](int k0, AU& a, AU& b) {
        if (!rowOK) {
            const v4u z4 = {0u, 0u, 0u, 0u};
            a.u[0] = z4; a.u[1] = z4; a.u[2] = z4; a.u[3] = z4;
        } else if (MODE == 0) {
            const v4u* as = (const v4u*)(Abf + (size_t)grow * K + (k0 + acseg));
            a.u[0] = as[0]; a.u[1] = as[1]; a.u[2] = as[2]; a.u[3] = as[3];
        } else {
            gather16(k0 + acseg,      a.s);
            gather16(k0 + acseg + 16, a.s + 16);
        }
        const v4u* bs = (const v4u*)(Wtbf + (size_t)(n0 + bn) * K + k0 + bkseg);
        b.u[0] = bs[0]; b.u[1] = bs[1]; b.u[2] = bs[2]; b.u[3] = bs[3];
    };

    auto storeAB = [&](int buf, const AU& a, const AU& b) {
        unsigned short* ad = &ldsA[buf][arow * LROW + acseg];
        *(v4u*)(ad)      = a.u[0]; *(v4u*)(ad + 8)  = a.u[1];
        *(v4u*)(ad + 16) = a.u[2]; *(v4u*)(ad + 24) = a.u[3];
        unsigned short* bd = &ldsB[buf][bn * LROW + bkseg];
        *(v4u*)(bd)      = b.u[0]; *(v4u*)(bd + 8)  = b.u[1];
        *(v4u*)(bd + 16) = b.u[2]; *(v4u*)(bd + 24) = b.u[3];
    };

    // prologue: stage first K-tile
    {
        AU a, b;
        loadAB(0, a, b);
        storeAB(0, a, b);
    }
    __syncthreads();

    int cur = 0;
    for (int k0 = 0; k0 < K; k0 += 64) {
        const bool hasNext = (k0 + 64) < K;     // uniform across block
        AU na, nb;
        if (hasNext) loadAB(k0 + 64, na, nb);   // prefetch into registers

        const unsigned short* lA = ldsA[cur];
        const unsigned short* lB = ldsB[cur];
        const int fr    = lane & 15;
        const int abase = (lane < 16) ? 0 : 8;
        const int bbase = (lane < 16) ? 0 : 8;
        const unsigned short* ap = &lA[(wave * 16 + fr) * LROW];

        union FU { v16bf v; v4u u[2]; };
        #pragma unroll
        for (int half = 0; half < 2; half++) {
            const int hk = half * 32;
            FU af;
            af.u[0] = *(const v4u*)(ap + hk + abase);        // K = abase..abase+7
            af.u[1] = *(const v4u*)(ap + hk + abase + 16);   // K = abase+16..+23
            // ping-pong B fragments: next fragment loads overlap current WMMA
            FU b0, b1;
            {
                const unsigned short* bp = &lB[fr * LROW + hk + bbase];
                b0.u[0] = *(const v4u*)bp; b0.u[1] = *(const v4u*)(bp + 16);
            }
            #pragma unroll
            for (int nt = 0; nt < 8; nt++) {
                FU& curB = (nt & 1) ? b1 : b0;
                FU& nxtB = (nt & 1) ? b0 : b1;
                if (nt < 7) {
                    const unsigned short* bp =
                        &lB[((nt + 1) * 16 + fr) * LROW + hk + bbase];
                    nxtB.u[0] = *(const v4u*)bp;
                    nxtB.u[1] = *(const v4u*)(bp + 16);
                }
                acc[nt] = __builtin_amdgcn_wmma_f32_16x16x32_bf16(
                    false, af.v, false, curB.v, (short)0, acc[nt], false, false);
            }
        }

        if (hasNext) {
            storeAB(cur ^ 1, na, nb);
            __syncthreads();
        }
        cur ^= 1;
    }

    // ---- epilogue: bias (+ residual), f32 store ----
    const int ccol  = lane & 15;
    const int rbase = m0 + wave * 16 + ((lane < 16) ? 0 : 8);
    #pragma unroll
    for (int nt = 0; nt < 8; nt++) {
        const int gc = n0 + nt * 16 + ccol;
        const float bv = bias[gc];
        #pragma unroll
        for (int vv = 0; vv < 8; vv++) {
            const int gr = rbase + vv;
            if (gr < M) {
                float val = acc[nt][vv] + bv;
                if (resid) val += resid[(size_t)gr * HDIM + gc];
                Cout[(size_t)gr * HDIM + gc] = val;
            }
        }
    }
}

extern "C" void kernel_launch(void* const* d_in, const int* in_sizes, int n_in,
                              void* d_out, int out_size, void* d_ws, size_t ws_size,
                              hipStream_t stream) {
    const float* x    = (const float*)d_in[0];
    const float* e    = (const float*)d_in[1];
    const int*   eidx = (const int*)d_in[2];
    const int N = in_sizes[0] / HDIM;
    const int E = in_sizes[1] / HDIM;

    const float* nW0 = (const float*)d_in[3];  const float* nb0 = (const float*)d_in[4];
    const float* ng0 = (const float*)d_in[5];  const float* nB0 = (const float*)d_in[6];
    const float* nW1 = (const float*)d_in[7];  const float* nb1 = (const float*)d_in[8];
    const float* ng1 = (const float*)d_in[9];  const float* nB1 = (const float*)d_in[10];
    const float* nW2 = (const float*)d_in[11]; const float* nb2 = (const float*)d_in[12];
    const float* eW0 = (const float*)d_in[13]; const float* eb0 = (const float*)d_in[14];
    const float* eg0 = (const float*)d_in[15]; const float* eB0 = (const float*)d_in[16];
    const float* eW1 = (const float*)d_in[17]; const float* eb1 = (const float*)d_in[18];
    const float* eg1 = (const float*)d_in[19]; const float* eB1 = (const float*)d_in[20];
    const float* eW2 = (const float*)d_in[21]; const float* eb2 = (const float*)d_in[22];

    float* xout = (float*)d_out;                    // [N,512]
    float* eout = (float*)d_out + (size_t)N * HDIM; // [E,512]

    char* p = (char*)d_ws;
    float* msg = (float*)p;                    p += (size_t)N * HDIM * 4;
    float* tmp = (float*)p;                    p += (size_t)E * HDIM * 4;
    unsigned short* act = (unsigned short*)p;  p += (size_t)E * HDIM * 2;
    unsigned short* wn0 = (unsigned short*)p;  p += (size_t)2 * HDIM * HDIM * 2;
    unsigned short* wn1 = (unsigned short*)p;  p += (size_t)HDIM * HDIM * 2;
    unsigned short* wn2 = (unsigned short*)p;  p += (size_t)HDIM * HDIM * 2;
    unsigned short* we0 = (unsigned short*)p;  p += (size_t)3 * HDIM * HDIM * 2;
    unsigned short* we1 = (unsigned short*)p;  p += (size_t)HDIM * HDIM * 2;
    unsigned short* we2 = (unsigned short*)p;  p += (size_t)HDIM * HDIM * 2;

    hipMemsetAsync(msg, 0, (size_t)N * HDIM * 4, stream);

    scatter_add_kernel<<<(E * 128 + 255) / 256, 256, 0, stream>>>(e, eidx + E, msg, E);

    // weight transpose+convert: W[K,512] f32 -> Wt[512,K] bf16
    cvt_transpose_kernel<<<dim3(2 * HDIM / 32, 16), 256, 0, stream>>>(nW0, wn0, 2 * HDIM);
    cvt_transpose_kernel<<<dim3(HDIM / 32, 16), 256, 0, stream>>>(nW1, wn1, HDIM);
    cvt_transpose_kernel<<<dim3(HDIM / 32, 16), 256, 0, stream>>>(nW2, wn2, HDIM);
    cvt_transpose_kernel<<<dim3(3 * HDIM / 32, 16), 256, 0, stream>>>(eW0, we0, 3 * HDIM);
    cvt_transpose_kernel<<<dim3(HDIM / 32, 16), 256, 0, stream>>>(eW1, we1, HDIM);
    cvt_transpose_kernel<<<dim3(HDIM / 32, 16), 256, 0, stream>>>(eW2, we2, HDIM);

    dim3 blk(256);
    dim3 gN(4, (N + 127) / 128), gE(4, (E + 127) / 128);

    // ---- node MLP: concat(x, msg) -> 512 -> 512 -> 512 (+x) ----
    gemm512_kernel<2><<<gN, blk, 0, stream>>>(nullptr, x, msg, nullptr, nullptr,
                                              wn0, nb0, nullptr, tmp, N, 2 * HDIM);
    ln_relu_kernel<<<N, blk, 0, stream>>>(tmp, ng0, nB0, act);
    gemm512_kernel<0><<<gN, blk, 0, stream>>>(act, nullptr, nullptr, nullptr, nullptr,
                                              wn1, nb1, nullptr, tmp, N, HDIM);
    ln_relu_kernel<<<N, blk, 0, stream>>>(tmp, ng1, nB1, act);
    gemm512_kernel<0><<<gN, blk, 0, stream>>>(act, nullptr, nullptr, nullptr, nullptr,
                                              wn2, nb2, x, xout, N, HDIM);

    // ---- edge MLP: concat(x_out[start], x_out[end], e) -> 512 -> 512 -> 512 (+e) ----
    gemm512_kernel<1><<<gE, blk, 0, stream>>>(nullptr, xout, e, eidx, eidx + E,
                                              we0, eb0, nullptr, tmp, E, 3 * HDIM);
    ln_relu_kernel<<<E, blk, 0, stream>>>(tmp, eg0, eB0, act);
    gemm512_kernel<0><<<gE, blk, 0, stream>>>(act, nullptr, nullptr, nullptr, nullptr,
                                              we1, eb1, nullptr, tmp, E, HDIM);
    ln_relu_kernel<<<E, blk, 0, stream>>>(tmp, eg1, eB1, act);
    gemm512_kernel<0><<<gE, blk, 0, stream>>>(act, nullptr, nullptr, nullptr, nullptr,
                                              we2, eb2, e, eout, E, HDIM);
}
